// EdgePredictor_81630148428373
// MI455X (gfx1250) — matrix operand order repaired and checked by reading, not compiled, not run on hardware
//
#include <hip/hip_runtime.h>
#include <hip/hip_bf16.h>

typedef __attribute__((ext_vector_type(16))) _Float16 v16h;
typedef __attribute__((ext_vector_type(8)))  float    v8f;
typedef __attribute__((ext_vector_type(4)))  float    f4;

#define EMBED        128
#define NEDGE        500000
#define W1T_STRIDE   264   // halves per row (256 + 8 pad) -> 132 dwords, bank-conflict free
#define HBUF_STRIDE  132   // floats per row (128 + 4 pad) -> conflict free b128 reads
#define WAVES_PER_BLOCK 8
#define NBLOCKS      256

__launch_bounds__(256)
__global__ void edge_mlp_wmma(const float* __restrict__ node_embed,
                              const int*   __restrict__ edge_index,
                              const float* __restrict__ W1,
                              const float* __restrict__ b1,
                              const float* __restrict__ W2,
                              const float* __restrict__ b2,
                              float* __restrict__ out,
                              int nedges) {
    // ---- LDS: W1 transposed (f16, padded), W2 transposed, biases, per-wave h ----
    __shared__ _Float16 w1t[128 * W1T_STRIDE];                 // 67,584 B
    __shared__ float    w2t[2 * 128];                          // W2^T
    __shared__ float    b1s[128];
    __shared__ float    b2s[2];
    __shared__ float    hbuf[WAVES_PER_BLOCK][16 * HBUF_STRIDE]; // 67,584 B

    const int tid = threadIdx.x;

    // Stage W1 (row-major [256][128] f32) -> w1t[n][k] f16, padded stride.
    for (int idx = tid; idx < 256 * 128; idx += blockDim.x) {
        int k = idx >> 7;          // 0..255
        int n = idx & 127;         // 0..127
        w1t[n * W1T_STRIDE + k] = (_Float16)W1[idx];
    }
    // Stage W2 ([128][2]) -> w2t[c][j]
    for (int idx = tid; idx < 256; idx += blockDim.x) {
        int j = idx >> 1, c = idx & 1;
        w2t[c * 128 + j] = W2[j * 2 + c];
    }
    if (tid < 128) b1s[tid] = b1[tid];
    if (tid < 2)   b2s[tid] = b2[tid];
    __syncthreads();

    const int lane  = tid & 31;
    const int wave  = tid >> 5;
    const int m     = lane & 15;          // row within 16-edge tile / column within N-tile
    const bool hi   = lane >= 16;
    const int gwave = blockIdx.x * WAVES_PER_BLOCK + wave;
    const int total_waves = gridDim.x * WAVES_PER_BLOCK;
    float* hb = hbuf[wave];

    // Hoist per-lane bias values (b1[16*nt + m]) and class bias.
    float bias[8];
    #pragma unroll
    for (int nt = 0; nt < 8; ++nt) bias[nt] = b1s[16 * nt + m];
    const int   cls  = hi ? 1 : 0;
    const float outb = b2s[cls];
    const float* w2row = &w2t[cls * 128];

    const int ntiles = nedges >> 5;       // 32 edges per super-tile (exact: 500000/32)

    for (int t = gwave; t < ntiles; t += total_waves) {
        const int e0 = t * 32;
        // endpoint indices for the two 16-edge sub-tiles (redundant across half-waves)
        const int eA = e0 + m, eB = e0 + 16 + m;
        const int rA = edge_index[eA];
        const int cA = edge_index[nedges + eA];
        const int rB = edge_index[eB];
        const int cB = edge_index[nedges + eB];

        v8f acc0[8], acc1[8];
        v8f zero = {};
        #pragma unroll
        for (int i = 0; i < 8; ++i) { acc0[i] = zero; acc1[i] = zero; }

        // K = 256 (concat of two 128-vectors); 8 K-blocks of 32
        #pragma unroll
        for (int kb = 0; kb < 8; ++kb) {
            const int i0 = (kb < 4) ? rA : cA;
            const int i1 = (kb < 4) ? rB : cB;
            // A-layout (matches B): lanes 0-15 K 0..15, lanes 16-31 K 16..31
            const int koff = (kb & 3) * 32 + (hi ? 16 : 0);
            const float* p0 = node_embed + (long)i0 * EMBED + koff;
            const float* p1 = node_embed + (long)i1 * EMBED + koff;
            f4 a0f0 = *(const f4*)(p0 +  0);
            f4 a0f1 = *(const f4*)(p0 +  4);
            f4 a0f2 = *(const f4*)(p0 +  8);
            f4 a0f3 = *(const f4*)(p0 + 12);
            f4 a1f0 = *(const f4*)(p1 +  0);
            f4 a1f1 = *(const f4*)(p1 +  4);
            f4 a1f2 = *(const f4*)(p1 +  8);
            f4 a1f3 = *(const f4*)(p1 + 12);
            v16h a0, a1;
            a0[ 0]=(_Float16)a0f0.x; a0[ 1]=(_Float16)a0f0.y; a0[ 2]=(_Float16)a0f0.z; a0[ 3]=(_Float16)a0f0.w;
            a0[ 4]=(_Float16)a0f1.x; a0[ 5]=(_Float16)a0f1.y; a0[ 6]=(_Float16)a0f1.z; a0[ 7]=(_Float16)a0f1.w;
            a0[ 8]=(_Float16)a0f2.x; a0[ 9]=(_Float16)a0f2.y; a0[10]=(_Float16)a0f2.z; a0[11]=(_Float16)a0f2.w;
            a0[12]=(_Float16)a0f3.x; a0[13]=(_Float16)a0f3.y; a0[14]=(_Float16)a0f3.z; a0[15]=(_Float16)a0f3.w;
            a1[ 0]=(_Float16)a1f0.x; a1[ 1]=(_Float16)a1f0.y; a1[ 2]=(_Float16)a1f0.z; a1[ 3]=(_Float16)a1f0.w;
            a1[ 4]=(_Float16)a1f1.x; a1[ 5]=(_Float16)a1f1.y; a1[ 6]=(_Float16)a1f1.z; a1[ 7]=(_Float16)a1f1.w;
            a1[ 8]=(_Float16)a1f2.x; a1[ 9]=(_Float16)a1f2.y; a1[10]=(_Float16)a1f2.z; a1[11]=(_Float16)a1f2.w;
            a1[12]=(_Float16)a1f3.x; a1[13]=(_Float16)a1f3.y; a1[14]=(_Float16)a1f3.z; a1[15]=(_Float16)a1f3.w;

            // B fragments: column n = 16*nt + m of W1, K-slice; lanes 0-15 K 0..15, hi lanes K 16..31
            #pragma unroll
            for (int nt = 0; nt < 8; ++nt) {
                const _Float16* bp =
                    &w1t[(16 * nt + m) * W1T_STRIDE + kb * 32 + (hi ? 16 : 0)];
                v16h bf = *(const v16h*)bp;
                acc0[nt] = __builtin_amdgcn_wmma_f32_16x16x32_f16(
                    false, a0, false, bf, (short)0, acc0[nt], false, false);
                acc1[nt] = __builtin_amdgcn_wmma_f32_16x16x32_f16(
                    false, a1, false, bf, (short)0, acc1[nt], false, false);
            }
        }

        // Bias + ReLU + layer 2, one 16-edge sub-tile at a time (h buffer reused)
        #pragma unroll
        for (int s = 0; s < 2; ++s) {
            v8f* acc = s ? acc1 : acc0;
            // C/D layout: vgpr r, lanes 0-15 -> (M=r, N=m), lanes 16-31 -> (M=r+8, N=m)
            #pragma unroll
            for (int nt = 0; nt < 8; ++nt) {
                v8f v = acc[nt];
                float bb = bias[nt];
                #pragma unroll
                for (int r = 0; r < 8; ++r) {
                    float x = v[r] + bb;
                    x = x > 0.0f ? x : 0.0f;
                    hb[(r + (hi ? 8 : 0)) * HBUF_STRIDE + 16 * nt + m] = x;
                }
            }
            // same-wave cross-lane LDS visibility; block compiler reordering
            asm volatile("s_wait_dscnt 0" ::: "memory");

            // Layer 2: lane -> (edge = m, class = cls); 128-deep f32 dot
            const float* hrow = &hb[m * HBUF_STRIDE];
            float sum = outb;
            #pragma unroll
            for (int j = 0; j < 128; j += 8) {
                f4 h0 = *(const f4*)(hrow + j);
                f4 h1 = *(const f4*)(hrow + j + 4);
                f4 w0 = *(const f4*)(w2row + j);
                f4 w1v = *(const f4*)(w2row + j + 4);
                sum = fmaf(h0.x, w0.x, sum);
                sum = fmaf(h0.y, w0.y, sum);
                sum = fmaf(h0.z, w0.z, sum);
                sum = fmaf(h0.w, w0.w, sum);
                sum = fmaf(h1.x, w1v.x, sum);
                sum = fmaf(h1.y, w1v.y, sum);
                sum = fmaf(h1.z, w1v.z, sum);
                sum = fmaf(h1.w, w1v.w, sum);
            }
            out[(e0 + s * 16 + m) * 2 + cls] = sum;
            asm volatile("s_wait_dscnt 0" ::: "memory"); // WAR guard before next sub-tile's stores
        }
    }
}

extern "C" void kernel_launch(void* const* d_in, const int* in_sizes, int n_in,
                              void* d_out, int out_size, void* d_ws, size_t ws_size,
                              hipStream_t stream) {
    const float* node_embed = (const float*)d_in[0];
    const int*   edge_index = (const int*)d_in[1];
    const float* W1         = (const float*)d_in[2];
    const float* b1         = (const float*)d_in[3];
    const float* W2         = (const float*)d_in[4];
    const float* b2         = (const float*)d_in[5];
    float*       out        = (float*)d_out;

    const int nedges = in_sizes[1] / 2;   // edge_index is [2, E]

    hipLaunchKernelGGL(edge_mlp_wmma, dim3(NBLOCKS), dim3(256), 0, stream,
                       node_embed, edge_index, W1, b1, W2, b2, out, nedges);
}
